// GATLayer_44220983279640
// MI455X (gfx1250) — matrix operand order repaired and checked by reading, not compiled, not run on hardware
//
#include <hip/hip_runtime.h>

// ---- problem constants (from reference) ----
#define BB   16
#define DD   512
#define FIN  128
#define FOUT 128
#define HH   8
#define BD   (BB*DD)          // 8192 rows
#define NEGINF (-9.0e15f)

typedef __attribute__((ext_vector_type(2)))  float  v2f;
typedef __attribute__((ext_vector_type(8)))  float  v8f;
typedef __attribute__((ext_vector_type(16))) __bf16 v16bf;

struct U128x2 { uint4 lo, hi; };

// =====================================================================
// Kernel 1: h = x @ W  via v_wmma_f32_16x16x4_f32 (full f32 precision).
// Block = one 16-row M tile; wave w = N-tile w (16 cols). K loop = 32 WMMAs.
// Epilogue stores h TRANSPOSED: hT[b][f][j]  (f32 + bf16 copies).
// =====================================================================
__global__ __launch_bounds__(256) void gemm_xw(const float* __restrict__ x,
                                               const float* __restrict__ W,
                                               float* __restrict__ hTf,
                                               __bf16* __restrict__ hTb) {
    const int tile    = blockIdx.x;          // 0..511
    const int rowBase = tile * 16;           // global row (b*512 + j)
    const int wave    = threadIdx.x >> 5;    // N tile
    const int lane    = threadIdx.x & 31;
    const int m       = lane & 15;
    const int half    = lane >> 4;           // 0: K{0,1} / 1: K{2,3}
    const int col     = wave * 16 + m;       // B-operand column (n = lane&15)

    const float* xrow = x + (size_t)(rowBase + m) * FIN;
    v8f c = {};
#pragma unroll 4
    for (int kb = 0; kb < FIN; kb += 4) {
        // A (16x4 f32): lanes 0-15 -> K0/K1, lanes 16-31 -> K2/K3
        v2f a;
        a.x = xrow[kb + half * 2 + 0];
        a.y = xrow[kb + half * 2 + 1];
        // B (4x16 f32): mirrored layout, column = n per lane
        v2f b;
        b.x = W[(kb + half * 2 + 0) * FOUT + col];
        b.y = W[(kb + half * 2 + 1) * FOUT + col];
        c = __builtin_amdgcn_wmma_f32_16x16x4_f32(false, a, false, b,
                                                  (short)0, c, false, false);
    }
    // C/D layout: VGPR r, lane -> row (r + half*8), col (wave*16 + lane&15)
    const int bidx  = rowBase / DD;
    const int jbase = rowBase % DD;
    float*  fo = hTf + (size_t)bidx * FOUT * DD + (size_t)col * DD;
    __bf16* bo = hTb + (size_t)bidx * FOUT * DD + (size_t)col * DD;
#pragma unroll
    for (int r = 0; r < 8; ++r) {
        const int j = jbase + r + half * 8;
        const float v = c[r];
        fo[j] = v;
        bo[j] = (__bf16)v;
    }
}

// =====================================================================
// Kernel 2: s1[b,h,j] = h[b,j,:]·a1[h] + a_b[h],  s2[b,h,j] = h[b,j,:]·a2[h]
// One thread per (b,j); coalesced reads of hT; a_w via scalar loads.
// =====================================================================
__global__ __launch_bounds__(256) void s_kernel(const float* __restrict__ hTf,
                                                const float* __restrict__ a_w,
                                                const float* __restrict__ a_b,
                                                float* __restrict__ s1,
                                                float* __restrict__ s2) {
    const int idx = blockIdx.x * 256 + threadIdx.x;   // 0..8191
    const int b = idx >> 9, j = idx & 511;
    const float* hb = hTf + (size_t)b * FOUT * DD + j;
    float a1[HH] = {}, a2[HH] = {};
    for (int f = 0; f < FOUT; ++f) {
        const float v = hb[(size_t)f * DD];
#pragma unroll
        for (int h = 0; h < HH; ++h) {
            a1[h] += v * a_w[h * (2 * FOUT) + f];
            a2[h] += v * a_w[h * (2 * FOUT) + FOUT + f];
        }
    }
#pragma unroll
    for (int h = 0; h < HH; ++h) {
        s1[((size_t)b * HH + h) * DD + j] = a1[h] + a_b[h];
        s2[((size_t)b * HH + h) * DD + j] = a2[h];
    }
}

// =====================================================================
// Kernel 3: fused masked-softmax attention (flash style).
// Block = (b, 16-row i-tile). Wave = head. j in steps of 32 (=WMMA K).
// Scores computed directly in the bf16 A-operand lane layout.
// V slab (128 f x 32 j bf16) DOUBLE-BUFFERED in LDS via
// GLOBAL_LOAD_ASYNC_TO_LDS_B128 (asynccnt), prefetched one step ahead so the
// copy overlaps the previous step's VALU + v_wmma_f32_16x16x32_bf16 chain.
// =====================================================================
#define SLAB_STRIDE 40                   // bf16/row (80B: 16B aligned, bank-spread)
#define SLAB_ELEMS  (FOUT * SLAB_STRIDE) // per buffer
#define NSTEPS      (DD / 32)            // 16 j-steps

__global__ __launch_bounds__(256) void gat_attn(const float* __restrict__ s1,
                                                const float* __restrict__ s2,
                                                const int*   __restrict__ adj,
                                                const __bf16* __restrict__ hTb,
                                                float* __restrict__ out) {
    __shared__ __align__(16) __bf16 slab[2 * SLAB_ELEMS];     // 2 x 10 KB
    __shared__ float red[16 * FOUT];                          // 8 KB

    const int blk   = blockIdx.x;         // 512 blocks
    const int b     = blk >> 5;
    const int ibase = (blk & 31) * 16;
    const int head  = threadIdx.x >> 5;   // wave id
    const int lane  = threadIdx.x & 31;
    const int m     = lane & 15;          // matrix row (A) / column n (B)
    const int koff  = (lane & 16) >> 1;   // 0 or 8: C-layout row offset of this half-wave
    const int jo    = lane & 16;          // B-operand K base (0 or 16)

    const float s1v      = s1[((size_t)b * HH + head) * DD + ibase + m];
    const float* s2h     = s2 + ((size_t)b * HH + head) * DD;
    const int*   adjrow  = adj + ((size_t)b * DD + ibase + m) * DD;
    const __bf16* hTb_b  = hTb + (size_t)b * FOUT * DD;

    // this thread's 32B share of each slab: row f, 32B half (two b128 asyncs)
    const int sf = threadIdx.x >> 1;          // 0..127
    const int sh = threadIdx.x & 1;           // 0/1
    const __bf16* gsrc0 = hTb_b + (size_t)sf * DD + sh * 16;

    auto issue_slab = [&](int buf, int jb) {
        const __bf16* g = gsrc0 + jb;
        const unsigned ds = (unsigned)(uintptr_t)(slab + buf * SLAB_ELEMS +
                                                  sf * SLAB_STRIDE + sh * 16);
        asm volatile(
            "global_load_async_to_lds_b128 %0, %1, off\n\t"
            "global_load_async_to_lds_b128 %0, %1, off offset:16"
            :: "v"(ds), "v"(g) : "memory");
    };

    float mrun = NEGINF;   // matches reference: all-masked rows -> uniform
    float lrun = 0.0f;
    v8f acc[8] = {};

    for (int t = threadIdx.x; t < 16 * FOUT; t += 256) red[t] = 0.0f;

    issue_slab(0, 0);                       // prologue prefetch

    for (int it = 0; it < NSTEPS; ++it) {
        const int jb = it * 32;
        const __bf16* cur = slab + (it & 1) * SLAB_ELEMS;

        if (it + 1 < NSTEPS) {
            issue_slab((it + 1) & 1, jb + 32);     // prefetch next step
            asm volatile("s_wait_asynccnt 0x2" ::: "memory");  // oldest 2 done
        } else {
            asm volatile("s_wait_asynccnt 0x0" ::: "memory");
        }
        __syncthreads();   // all waves' current-buffer asyncs complete

        // ---- 16 scores per lane, directly in A-operand k-order ----
        float sc[16];
        float tmax = -3.0e38f;
#pragma unroll
        for (int t = 0; t < 16; ++t) {
            const int k = t + jo;                            // {0..15}(+16)
            float e = s1v + s2h[jb + k];
            e = (e >= 0.0f) ? e : 0.01f * e;                 // leaky_relu
            sc[t] = (adjrow[jb + k] > 0) ? e : NEGINF;
            tmax = fmaxf(tmax, sc[t]);
        }
        tmax = fmaxf(tmax, __shfl_xor(tmax, 16, 32));        // both k-halves of row
        const float mnew  = fmaxf(mrun, tmax);
        const float alpha = __expf(mrun - mnew);

        v16bf aop;
        float psum = 0.0f;
#pragma unroll
        for (int t = 0; t < 16; ++t) {
            const float p = __expf(sc[t] - mnew);
            psum += p;
            aop[t] = (__bf16)p;
        }
        psum += __shfl_xor(psum, 16, 32);
        lrun = lrun * alpha + psum;
        mrun = mnew;

        // ---- rescale accumulators (alpha broadcast per C-layout row) ----
        float ar[8];
#pragma unroll
        for (int r = 0; r < 8; ++r) ar[r] = __shfl(alpha, r + koff, 32);
#pragma unroll
        for (int ft = 0; ft < 8; ++ft)
#pragma unroll
            for (int r = 0; r < 8; ++r) acc[ft][r] *= ar[r];

        // ---- PV: 8 WMMAs (one per 16-col f tile) ----
#pragma unroll
        for (int ft = 0; ft < 8; ++ft) {
            const U128x2* q = (const U128x2*)(cur + (ft * 16 + m) * SLAB_STRIDE + jo);
            const v16bf bop = __builtin_bit_cast(v16bf, *q);
            acc[ft] = __builtin_amdgcn_wmma_f32_16x16x32_bf16(false, aop, false, bop,
                                                              (short)0, acc[ft],
                                                              false, false);
        }
        __syncthreads();   // reads done before this buffer is overwritten (it+2)
    }

    // ---- normalize (1/l_i * 1/H) and reduce heads via LDS atomics ----
    const float inv = 0.125f / lrun;          // lrun >= 1 always (max term = 1)
    float linv[8];
#pragma unroll
    for (int r = 0; r < 8; ++r) linv[r] = __shfl(inv, r + koff, 32);
#pragma unroll
    for (int ft = 0; ft < 8; ++ft)
#pragma unroll
        for (int r = 0; r < 8; ++r) {
            const int rr = r + koff;                          // local row 0..15
            atomicAdd(&red[rr * FOUT + ft * 16 + m], acc[ft][r] * linv[r]);
        }
    __syncthreads();

    for (int t = threadIdx.x; t < 16 * FOUT; t += 256) {
        const int row = t >> 7, f = t & 127;
        out[((size_t)b * DD + ibase + row) * FOUT + f] = red[t];
    }
}

// =====================================================================
extern "C" void kernel_launch(void* const* d_in, const int* in_sizes, int n_in,
                              void* d_out, int out_size, void* d_ws, size_t ws_size,
                              hipStream_t stream) {
    const float* x   = (const float*)d_in[0];   // (16,512,128) f32
    const int*   adj = (const int*)  d_in[1];   // (16,512,512) i32
    const float* W   = (const float*)d_in[2];   // (128,128) f32
    const float* a_w = (const float*)d_in[3];   // (8,256) f32
    const float* a_b = (const float*)d_in[4];   // (8,) f32
    float* out = (float*)d_out;                 // (16,512,128) f32

    char* ws = (char*)d_ws;                     // needs ~6.6 MB
    float*  hTf = (float*)ws;                                  // 4 MB  [B][F][D] f32
    __bf16* hTb = (__bf16*)(ws + 4u * 1024 * 1024);            // 2 MB  [B][F][D] bf16
    float*  s1  = (float*)(ws + 6u * 1024 * 1024);             // 256 KB [B][H][D]
    float*  s2  = (float*)(ws + 6u * 1024 * 1024 + 256 * 1024);// 256 KB [B][H][D]

    gemm_xw <<<BD / 16, 256, 0, stream>>>(x, W, hTf, hTb);
    s_kernel<<<BD / 256, 256, 0, stream>>>(hTf, a_w, a_b, s1, s2);
    gat_attn<<<BB * (DD / 16), 256, 0, stream>>>(s1, s2, adj, hTb, out);
}
